// Many2ManyRNN_13529146982530
// MI455X (gfx1250) — matrix operand order repaired and checked by reading, not compiled, not run on hardware
//
#include <hip/hip_runtime.h>
#include <hip/hip_bf16.h>
#include <cstdint>

// Problem dims (match reference)
#define B_ 256
#define S_ 512
#define I_ 256
#define H_ 256
#define O_ 256
#define L_ 64

// Padded LDS row stride (bf16 elems) for 256-wide tiles: 264 -> 528B rows,
// shifts banks by 4 per row -> conflict-free A-fragment reads.
#define LDSP 264

typedef __attribute__((ext_vector_type(16))) __bf16 v16bf;
typedef __attribute__((ext_vector_type(8)))  __bf16 v8bf;
typedef __attribute__((ext_vector_type(8)))  float  v8f;
typedef __attribute__((ext_vector_type(4)))  float  v4f;

static __device__ __forceinline__ v16bf combine16(v8bf lo, v8bf hi) {
  return __builtin_shufflevector(lo, hi, 0,1,2,3,4,5,6,7,8,9,10,11,12,13,14,15);
}

// A-matrix fragment (16x32 bf16, M x K). pA = row/half base (lane-invariant
// inside the k-loop): pA = sA + (lane&15)*LDSP + 8*(lane>>4). Offsets become
// ds_load immediates.
static __device__ __forceinline__ v16bf a_frag(const __bf16* pA, int k0) {
  v8bf lo = *(const v8bf*)(pA + k0);
  v8bf hi = *(const v8bf*)(pA + k0 + 16);
  return combine16(lo, hi);
}

static __device__ __forceinline__ v8f wmma_bf16(v16bf a, v16bf b, v8f c) {
  // D = A x B + C, f32 accumulate
  return __builtin_amdgcn_wmma_f32_16x16x32_bf16(false, a, false, b, (short)0, c,
                                                 false, false);
}

static __device__ __forceinline__ float sigmoidf_(float x) {
  return 1.0f / (1.0f + __expf(-x));
}

// ---------------------------------------------------------------------------
// Kernel 0: reformat the five 256x256 f32 weight matrices into bf16
// B-fragment order. elem index e = ((nt*8 + kc)*32 + lane)*16 + j maps to
// W[k][n] with n = nt*16 + (lane&15), k = kc*32 + (lane>>4)*8 + (j>>3)*16 + (j&7).
// ---------------------------------------------------------------------------
__global__ void __launch_bounds__(256)
prep_frag_kernel(const float* __restrict__ W1x, const float* __restrict__ W1a,
                 const float* __restrict__ W2x, const float* __restrict__ W2a,
                 const float* __restrict__ Wy, __bf16* __restrict__ wsW) {
  const float* srcs[5] = {W1x, W1a, W2x, W2a, Wy};
  int mat = blockIdx.y;
  const float* W = srcs[mat];
  __bf16* dst = wsW + (size_t)mat * (256 * 256);
  int e = blockIdx.x * 256 + threadIdx.x;     // 0..65535
  int j    =  e        & 15;
  int lane = (e >> 4)  & 31;
  int kc   = (e >> 9)  & 7;
  int nt   =  e >> 12;
  int n = nt * 16 + (lane & 15);
  int k = kc * 32 + (lane >> 4) * 8 + (j >> 3) * 16 + (j & 7);
  dst[e] = (__bf16)W[k * 256 + n];
}

// ---------------------------------------------------------------------------
// Kernel 1: P[t,b,:] = x[b,t,:] @ W1x + (b1x + b1a), stored bf16 in C-fragment
// order: Pf[t*65536 + mtile*4096 + nt*256 + lane*8 + r].
// Grid: (S*B/128, H/64); block 256 (8 waves). Wave = 16 rows x 4 ntiles.
// All B-fragment loads share one per-lane base pointer + immediate offsets.
// ---------------------------------------------------------------------------
__global__ void __launch_bounds__(256)
xw_kernel(const float* __restrict__ x, const float* __restrict__ b1x,
          const float* __restrict__ b1a, const __bf16* __restrict__ W1xf,
          __bf16* __restrict__ Pf) {
  int lane = threadIdx.x & 31;
  int w    = threadIdx.x >> 5;
  int r0 = blockIdx.x * 128 + w * 16;   // global row = t*B + b  (16 rows share t)
  int t  = r0 >> 8;                     // B_ == 256
  int b0 = r0 & 255;
  int ntBase = blockIdx.y * 4;

  int m  = lane & 15;
  int kh = lane >> 4;
  const float* ax = x + ((size_t)(b0 + m) * S_ + t) * I_ + 8 * kh;

  // B-fragment base (v16bf units): index (nt*8+kc)*32 + lane
  const v16bf* pB = (const v16bf*)W1xf + (size_t)ntBase * 256 + lane;

  v8f c[4];
  #pragma unroll
  for (int q = 0; q < 4; ++q)
    #pragma unroll
    for (int r = 0; r < 8; ++r) c[q][r] = 0.0f;

  #pragma unroll
  for (int kc = 0; kc < 8; ++kc) {
    int k0 = kc * 32;
    // Assemble A fragment from global f32 -> bf16 (immediate offsets off ax)
    v4f a0 = *(const v4f*)(ax + k0);
    v4f a1 = *(const v4f*)(ax + k0 + 4);
    v4f a2 = *(const v4f*)(ax + k0 + 16);
    v4f a3 = *(const v4f*)(ax + k0 + 20);
    v16bf a;
    #pragma unroll
    for (int j = 0; j < 4; ++j) {
      a[j]      = (__bf16)a0[j];
      a[4 + j]  = (__bf16)a1[j];
      a[8 + j]  = (__bf16)a2[j];
      a[12 + j] = (__bf16)a3[j];
    }
    #pragma unroll
    for (int q = 0; q < 4; ++q)
      c[q] = wmma_bf16(a, pB[q * 256 + kc * 32], c[q]);
  }

  int mtile = b0 >> 4;
  __bf16* pOut = Pf + (((size_t)t * 16 + mtile) * 16 + ntBase) * 256 + lane * 8;
  const float* pb1x = b1x + ntBase * 16 + (lane & 15);
  const float* pb1a = b1a + ntBase * 16 + (lane & 15);
  #pragma unroll
  for (int q = 0; q < 4; ++q) {
    float bias = pb1x[q * 16] + pb1a[q * 16];
    v8bf pbv;
    #pragma unroll
    for (int r = 0; r < 8; ++r) pbv[r] = (__bf16)(c[q][r] + bias);
    *(v8bf*)(pOut + q * 256) = pbv;
  }
}

// ---------------------------------------------------------------------------
// Kernel 2: encoder recurrence. 16 WGs; WG g owns batch rows [16g,16g+16).
// a in LDS (bf16, padded). Per step: c = P_t + a@W1a (WMMA), tanh, write back.
// P fragments for step t+1 are double-buffered into LDS with
// GLOBAL_LOAD_ASYNC_TO_LDS_B128 (ASYNCcnt) while step t computes. Each lane
// async-writes exactly the 16B it later reads -> only s_wait_asynccnt needed.
// ---------------------------------------------------------------------------
__global__ void __launch_bounds__(256)
enc_rec_kernel(const __bf16* __restrict__ Pf, const __bf16* __restrict__ W1af,
               __bf16* __restrict__ aLast) {
  __shared__ __bf16 sA[16 * LDSP];
  __shared__ __bf16 sP[2][16 * 256];      // double-buffered P tile (8KB each)
  int tid  = threadIdx.x;
  int lane = tid & 31;
  int w    = tid >> 5;
  int g    = blockIdx.x;                  // batch tile
  int nt0 = 2 * w;
  int h  = lane >> 4;
  int nl = lane & 15;

  for (int i = tid; i < 16 * LDSP; i += 256) sA[i] = (__bf16)0.0f;
  __syncthreads();

  // Loop-invariant bases
  const v16bf*  pB = (const v16bf*)W1af + (size_t)nt0 * 256 + lane;       // B frags
  const __bf16* pA = sA + (lane & 15) * LDSP + 8 * (lane >> 4);           // A frags
  const __bf16* gP = Pf + (size_t)g * 4096 + (size_t)nt0 * 256 + lane * 8; // global P
  const __bf16* sPr = &sP[0][0] + nt0 * 256 + lane * 8;                   // LDS P read
  uint32_t lds0 = (uint32_t)(uintptr_t)(&sP[0][0]) + nt0 * 512 + lane * 16;

  // Prologue: stage P tile for t = 0 into buffer 0
  {
    const __bf16* gsrc = gP;
    asm volatile("global_load_async_to_lds_b128 %0, %1, off"
                 :: "v"(lds0), "v"(gsrc) : "memory");
    asm volatile("global_load_async_to_lds_b128 %0, %1, off offset:512"
                 :: "v"(lds0), "v"(gsrc) : "memory");
  }

  for (int t = 0; t < S_; ++t) {
    // Issue async staging of P for step t+1 (clamped; uniform control flow)
    {
      int tn = (t + 1 < S_) ? (t + 1) : (S_ - 1);
      const __bf16* gsrc = gP + (size_t)tn * 65536;
      uint32_t ldsn = lds0 + (((t + 1) & 1) << 13);   // +8192B per buffer
      asm volatile("global_load_async_to_lds_b128 %0, %1, off"
                   :: "v"(ldsn), "v"(gsrc) : "memory");
      asm volatile("global_load_async_to_lds_b128 %0, %1, off offset:512"
                   :: "v"(ldsn), "v"(gsrc) : "memory");
    }
    // Wait for this step's 2 staged loads (in-order; 2 newest may be pending)
    asm volatile("s_wait_asynccnt 0x2" ::: "memory");

    v8f c0, c1;
    {
      const __bf16* sp = sPr + (size_t)(t & 1) * (16 * 256);
      v8bf p0 = *(const v8bf*)(sp);
      v8bf p1 = *(const v8bf*)(sp + 256);
      #pragma unroll
      for (int r = 0; r < 8; ++r) { c0[r] = (float)p0[r]; c1[r] = (float)p1[r]; }
    }

    // Software-pipelined k-loop: B frags loaded one kc ahead
    v16bf bA = pB[0];
    v16bf bB = pB[256];
    #pragma unroll
    for (int kc = 0; kc < 8; ++kc) {
      v16bf bA1, bB1;
      if (kc < 7) { bA1 = pB[(kc + 1) * 32]; bB1 = pB[256 + (kc + 1) * 32]; }
      v16bf a = a_frag(pA, kc * 32);
      c0 = wmma_bf16(a, bA, c0);
      c1 = wmma_bf16(a, bB, c1);
      bA = bA1; bB = bB1;
    }

    #pragma unroll
    for (int r = 0; r < 8; ++r) { c0[r] = tanhf(c0[r]); c1[r] = tanhf(c1[r]); }
    __syncthreads();
    #pragma unroll
    for (int r = 0; r < 8; ++r) {
      sA[(8 * h + r) * LDSP + nt0 * 16 + nl]      = (__bf16)c0[r];
      sA[(8 * h + r) * LDSP + nt0 * 16 + 16 + nl] = (__bf16)c1[r];
    }
    __syncthreads();
  }

  for (int i = tid; i < 16 * 256; i += 256) {
    int row = i >> 8, col = i & 255;
    aLast[(size_t)(g * 16 + row) * 256 + col] = sA[row * LDSP + col];
  }
}

// ---------------------------------------------------------------------------
// Kernel 3: decoder. 16 WGs x 16 batch rows. y0 = sigmoid(a@Wy+by) (kept in
// LDS as next A operand); then 64 steps:
//   a = tanh(a@W2a + (i==0 ? y0@W2x : 0) + b2x + b2a)
//   y = sigmoid(a@Wy + by) -> out[b, i, :]
// ---------------------------------------------------------------------------
__global__ void __launch_bounds__(256)
dec_kernel(const __bf16* __restrict__ aLast, const __bf16* __restrict__ W2xf,
           const __bf16* __restrict__ W2af, const __bf16* __restrict__ Wyf,
           const float* __restrict__ b2x, const float* __restrict__ b2a,
           const float* __restrict__ by, float* __restrict__ out) {
  __shared__ __bf16 sA[16 * LDSP];
  __shared__ __bf16 sY[16 * LDSP];
  int tid = threadIdx.x, lane = tid & 31, w = tid >> 5;
  int g = blockIdx.x;
  int b0 = g * 16;
  int nt0 = 2 * w;
  int h = lane >> 4, nl = lane & 15;
  int n0 = nt0 * 16 + nl, n1 = n0 + 16;
  float by0 = by[n0], by1 = by[n1];
  float bh0 = b2x[n0] + b2a[n0];
  float bh1 = b2x[n1] + b2a[n1];

  // Loop-invariant fragment bases
  const v16bf* pW2a = (const v16bf*)W2af + (size_t)nt0 * 256 + lane;
  const v16bf* pW2x = (const v16bf*)W2xf + (size_t)nt0 * 256 + lane;
  const v16bf* pWy  = (const v16bf*)Wyf  + (size_t)nt0 * 256 + lane;
  const __bf16* pA  = sA + (lane & 15) * LDSP + 8 * (lane >> 4);
  const __bf16* pAY = sY + (lane & 15) * LDSP + 8 * (lane >> 4);

  for (int i = tid; i < 16 * 256; i += 256) {
    int row = i >> 8, col = i & 255;
    sA[row * LDSP + col] = aLast[(size_t)(b0 + row) * 256 + col];
  }
  __syncthreads();

  // y0 = sigmoid(a_last @ Wy + by) -> sY
  {
    v8f c0, c1;
    #pragma unroll
    for (int r = 0; r < 8; ++r) { c0[r] = 0.0f; c1[r] = 0.0f; }
    #pragma unroll
    for (int kc = 0; kc < 8; ++kc) {
      v16bf a = a_frag(pA, kc * 32);
      c0 = wmma_bf16(a, pWy[kc * 32], c0);
      c1 = wmma_bf16(a, pWy[256 + kc * 32], c1);
    }
    __syncthreads();
    #pragma unroll
    for (int r = 0; r < 8; ++r) {
      sY[(8 * h + r) * LDSP + n0] = (__bf16)sigmoidf_(c0[r] + by0);
      sY[(8 * h + r) * LDSP + n1] = (__bf16)sigmoidf_(c1[r] + by1);
    }
    __syncthreads();
  }

  for (int i = 0; i < L_; ++i) {
    // hidden update
    v8f c0, c1;
    #pragma unroll
    for (int r = 0; r < 8; ++r) { c0[r] = bh0; c1[r] = bh1; }
    #pragma unroll
    for (int kc = 0; kc < 8; ++kc) {
      v16bf a = a_frag(pA, kc * 32);
      c0 = wmma_bf16(a, pW2a[kc * 32], c0);
      c1 = wmma_bf16(a, pW2a[256 + kc * 32], c1);
    }
    if (i == 0) {   // uniform branch: EXEC stays full for WMMA
      #pragma unroll
      for (int kc = 0; kc < 8; ++kc) {
        v16bf a = a_frag(pAY, kc * 32);
        c0 = wmma_bf16(a, pW2x[kc * 32], c0);
        c1 = wmma_bf16(a, pW2x[256 + kc * 32], c1);
      }
    }
    #pragma unroll
    for (int r = 0; r < 8; ++r) { c0[r] = tanhf(c0[r]); c1[r] = tanhf(c1[r]); }
    __syncthreads();   // all waves done reading sA (and sY) for this step
    #pragma unroll
    for (int r = 0; r < 8; ++r) {
      sA[(8 * h + r) * LDSP + n0] = (__bf16)c0[r];
      sA[(8 * h + r) * LDSP + n1] = (__bf16)c1[r];
    }
    __syncthreads();

    // y_i = sigmoid(a @ Wy + by) -> out[b, i, :]
    v8f d0, d1;
    #pragma unroll
    for (int r = 0; r < 8; ++r) { d0[r] = by0; d1[r] = by1; }
    #pragma unroll
    for (int kc = 0; kc < 8; ++kc) {
      v16bf a = a_frag(pA, kc * 32);
      d0 = wmma_bf16(a, pWy[kc * 32], d0);
      d1 = wmma_bf16(a, pWy[256 + kc * 32], d1);
    }
    #pragma unroll
    for (int r = 0; r < 8; ++r) {
      int row = b0 + 8 * h + r;
      out[((size_t)row * L_ + i) * O_ + n0] = sigmoidf_(d0[r]);
      out[((size_t)row * L_ + i) * O_ + n1] = sigmoidf_(d1[r]);
    }
    // next iteration's sA write is guarded by the barrier before the write
  }
}

// ---------------------------------------------------------------------------
extern "C" void kernel_launch(void* const* d_in, const int* in_sizes, int n_in,
                              void* d_out, int out_size, void* d_ws, size_t ws_size,
                              hipStream_t stream) {
  (void)in_sizes; (void)n_in; (void)out_size; (void)ws_size;
  const float* x   = (const float*)d_in[0];
  const float* W1x = (const float*)d_in[1];
  const float* b1x = (const float*)d_in[2];
  const float* W1a = (const float*)d_in[3];
  const float* b1a = (const float*)d_in[4];
  const float* W2x = (const float*)d_in[5];
  const float* b2x = (const float*)d_in[6];
  const float* W2a = (const float*)d_in[7];
  const float* b2a = (const float*)d_in[8];
  const float* Wy  = (const float*)d_in[9];
  const float* by  = (const float*)d_in[10];
  float* out = (float*)d_out;

  // Workspace layout (bf16 elems):
  //   [0 .. 5*65536)         : W1x,W1a,W2x,W2a,Wy in B-fragment order
  //   [5*65536 .. 6*65536)   : a_last (row-major [B][H])
  //   [6*65536 .. +S*B*H)    : P in C-fragment order (64 MB)
  __bf16* ws    = (__bf16*)d_ws;
  __bf16* Wf    = ws;
  __bf16* aLast = ws + (size_t)5 * 65536;
  __bf16* Pf    = ws + (size_t)6 * 65536;

  prep_frag_kernel<<<dim3(256, 5), 256, 0, stream>>>(W1x, W1a, W2x, W2a, Wy, Wf);
  xw_kernel<<<dim3((S_ * B_) / 128, H_ / 64), 256, 0, stream>>>(x, b1x, b1a, Wf, Pf);
  enc_rec_kernel<<<16, 256, 0, stream>>>(Pf, Wf + (size_t)1 * 65536, aLast);
  dec_kernel<<<16, 256, 0, stream>>>(aLast, Wf + (size_t)2 * 65536,
                                     Wf + (size_t)3 * 65536, Wf + (size_t)4 * 65536,
                                     b2x, b2a, by, out);
}
